// MinLSTM_8160437862912
// MI455X (gfx1250) — compile-verified, matrix operand order checked by reading, not run on hardware
//
#include <hip/hip_runtime.h>
#include <hip/hip_bf16.h>

// ---------------------------------------------------------------------------
// minLSTM forward for MI455X (gfx1250): f16 WMMA gate GEMMs + chunked log-scan
// ---------------------------------------------------------------------------

constexpr int kB = 16;
constexpr int kT = 2048;
constexpr int kE = 512;            // embedding dim (== K of layer 0)
constexpr int kH = 512;
constexpr int kM = 512;
constexpr int kN = kB * kT;        // 32768 GEMM rows
constexpr int kK = 512;            // inner dim (both layers)

constexpr int BM = 128;            // block tile rows (8 waves x 16)
constexpr int BN = 64;             // block tile cols
constexpr int BK = 32;             // K step (== WMMA K)
constexpr int LPAD = 8;            // LDS row pad (halves) -> 80B pitch, conflict-free b128

constexpr int CHUNK = 64;          // scan chunk length
constexpr int NCH = kT / CHUNK;    // 32 chunks

typedef __attribute__((ext_vector_type(16))) _Float16 v16h;
typedef __attribute__((ext_vector_type(8)))  _Float16 v8h;
typedef __attribute__((ext_vector_type(8)))  float    v8f;

__device__ __forceinline__ float softplus_f(float x) {
  if (x > 20.0f)  return x;
  if (x < -20.0f) return __expf(x);
  return log1pf(__expf(x));
}

// stable log(exp(a)+exp(b)); works with a == -inf
__device__ __forceinline__ float lse2(float a, float b) {
  float mx = fmaxf(a, b), mn = fminf(a, b);
  return mx + log1pf(__expf(mn - mx));
}

// ---------------------------------------------------------------------------
// Fused 3-gate GEMM:  [f|i|th] = A @ [Wf|Wi|Wh] + b, then gate math ->
// log_f, log_v = log_i + log_g(th).  A is f32 in global (optionally token-
// indirected through emb for layer 0), converted to f16 while staging to LDS.
// ---------------------------------------------------------------------------
__global__ __launch_bounds__(256)
void gate_gemm_kernel(const float* __restrict__ Asrc,   // (rows,kK) f32, or emb (A,kE)
                      const int*   __restrict__ tok,    // nullptr, or (kN,) token ids
                      const float* __restrict__ Wf, const float* __restrict__ Wi,
                      const float* __restrict__ Wh,
                      const float* __restrict__ bf, const float* __restrict__ bi,
                      const float* __restrict__ bh,
                      float* __restrict__ out_logf,     // (kN,kH)
                      float* __restrict__ out_logv)     // (kN,kH)
{
  __shared__ __align__(16) _Float16 Alds[BM][BK + LPAD];
  __shared__ __align__(16) _Float16 Blds[3][BN][BK + LPAD];   // transposed: [n][k]

  const int tid  = threadIdx.x;
  const int lane = tid & 31;
  const int wave = tid >> 5;
  const int blockRow = blockIdx.x * BM;
  const int blockCol = blockIdx.y * BN;

  // accumulators: 3 gates x 4 column tiles of 16x16 f32
  v8f acc[3][BN / 16];
#pragma unroll
  for (int g = 0; g < 3; ++g)
#pragma unroll
    for (int nt = 0; nt < BN / 16; ++nt)
#pragma unroll
      for (int j = 0; j < 8; ++j) acc[g][nt][j] = 0.0f;

  // --- staging maps ---
  // A: 128 rows x 32 k per step; each thread loads 16 consecutive f32
  const int arow = tid >> 1;
  const int acol = (tid & 1) * 16;
  const long arow_g = (long)blockRow + arow;
  const float* aptr = tok ? (Asrc + (long)tok[arow_g] * kK)
                          : (Asrc + arow_g * (long)kK);
  // B: per gate 32k x 64n; thread -> (kk4 = tid/64 in 0..3, nn = tid%64)
  const int nn  = tid & 63;
  const int kk4 = tid >> 6;

  // WMMA lane mapping (16-bit operands, wave32):
  //   row/col within tile = lane % 16 ; K half (0 or 8) = (lane/16)*8
  const int lm    = lane & 15;
  const int khalf = (lane >> 4) * 8;

  // B fragment loader (2x ds_load_b128)
  auto loadB = [&](int g, int nt) -> v16h {
    v8h lo = *reinterpret_cast<const v8h*>(&Blds[g][nt * 16 + lm][khalf]);
    v8h hi = *reinterpret_cast<const v8h*>(&Blds[g][nt * 16 + lm][khalf + 16]);
    v16h r;
#pragma unroll
    for (int j = 0; j < 8; ++j) { r[j] = lo[j]; r[8 + j] = hi[j]; }
    return r;
  };

  for (int k0 = 0; k0 < kK; k0 += BK) {
    // prefetch next A K-tile into caches (global_prefetch_b8)
    if (k0 + BK < kK) __builtin_prefetch(aptr + k0 + BK + acol, 0, 1);

    // stage A (f32 -> f16), two 16B ds_store_b128 per thread
    {
      const float4* ap4 = reinterpret_cast<const float4*>(aptr + k0 + acol);
      float4 t0 = ap4[0], t1 = ap4[1], t2 = ap4[2], t3 = ap4[3];
      v8h h0, h1;
      h0[0] = (_Float16)t0.x; h0[1] = (_Float16)t0.y;
      h0[2] = (_Float16)t0.z; h0[3] = (_Float16)t0.w;
      h0[4] = (_Float16)t1.x; h0[5] = (_Float16)t1.y;
      h0[6] = (_Float16)t1.z; h0[7] = (_Float16)t1.w;
      h1[0] = (_Float16)t2.x; h1[1] = (_Float16)t2.y;
      h1[2] = (_Float16)t2.z; h1[3] = (_Float16)t2.w;
      h1[4] = (_Float16)t3.x; h1[5] = (_Float16)t3.y;
      h1[6] = (_Float16)t3.z; h1[7] = (_Float16)t3.w;
      *reinterpret_cast<v8h*>(&Alds[arow][acol])     = h0;
      *reinterpret_cast<v8h*>(&Alds[arow][acol + 8]) = h1;
    }
    // stage B transposed: Blds[g][n][k] = W[k0+k][blockCol+n]; one b128 store
#pragma unroll
    for (int g = 0; g < 3; ++g) {
      const float* w = (g == 0) ? Wf : ((g == 1) ? Wi : Wh);
      v8h hb;
#pragma unroll
      for (int kk = 0; kk < 8; ++kk) {
        int k = kk4 * 8 + kk;
        hb[kk] = (_Float16)w[(long)(k0 + k) * kH + blockCol + nn];
      }
      *reinterpret_cast<v8h*>(&Blds[g][nn][kk4 * 8]) = hb;
    }
    __syncthreads();

    // A fragment for this wave's 16-row strip (2x ds_load_b128)
    v16h afrag;
    {
      v8h alo = *reinterpret_cast<const v8h*>(&Alds[wave * 16 + lm][khalf]);
      v8h ahi = *reinterpret_cast<const v8h*>(&Alds[wave * 16 + lm][khalf + 16]);
#pragma unroll
      for (int j = 0; j < 8; ++j) { afrag[j] = alo[j]; afrag[8 + j] = ahi[j]; }
    }

    // two-deep ping-pong of B fragments: WMMA on fragment idx overlaps with
    // the in-flight ds_load_b128 pair of fragment idx+2 (partial dscnt waits)
    v16h bbuf[2];
    bbuf[0] = loadB(0, 0);
    bbuf[1] = loadB(0, 1);
#pragma unroll
    for (int idx = 0; idx < 12; ++idx) {
      const int g  = idx >> 2;
      const int nt = idx & 3;
      v16h cur = bbuf[idx & 1];
      if (idx + 2 < 12) bbuf[idx & 1] = loadB((idx + 2) >> 2, (idx + 2) & 3);
      acc[g][nt] = __builtin_amdgcn_wmma_f32_16x16x32_f16(
          /*neg_a=*/false, afrag, /*neg_b=*/false, cur,
          /*c_mod=*/(short)0, acc[g][nt],
          /*reuse_a=*/false, /*reuse_b=*/false);
    }
    __syncthreads();
  }

  // epilogue: bias + gate math, write log_f and log_v
  // C/D layout: VGPR j -> M = j + 8*(lane/16), N = lane%16
  const int mbase = blockRow + wave * 16 + (lane >> 4) * 8;
#pragma unroll
  for (int nt = 0; nt < BN / 16; ++nt) {
    const int col = blockCol + nt * 16 + lm;
    const float vbf = bf[col], vbi = bi[col], vbh = bh[col];
#pragma unroll
    for (int j = 0; j < 8; ++j) {
      float fv = acc[0][nt][j] + vbf;
      float iv = acc[1][nt][j] + vbi;
      float tv = acc[2][nt][j] + vbh;
      float diff  = softplus_f(-fv) - softplus_f(-iv);
      float log_f = -softplus_f(diff);
      float log_i = -softplus_f(-diff);
      float log_th = (tv >= 0.0f) ? logf(tv + 0.5f) : -softplus_f(-tv);
      long row = (long)(mbase + j);
      out_logf[row * kH + col] = log_f;
      out_logv[row * kH + col] = log_i + log_th;
    }
  }
}

// ---------------------------------------------------------------------------
// Chunked log-space scan (matches cumsum + cumlogsumexp reference, incl. the
// implicit log h0 = 0 leading term).  State per channel: A = cumsum(log_f),
// L = logsumexp_{s<=t}(v_s - A_s);  h_t = exp(A + L).
// ---------------------------------------------------------------------------
__global__ __launch_bounds__(256)
void scan_pass1(const float* __restrict__ lf, const float* __restrict__ lv,
                float* __restrict__ sumA, float* __restrict__ sumL)
{
  const int tid = blockIdx.x * 256 + threadIdx.x;    // B*NCH*H threads
  const int d  = tid % kH;
  const int bc = tid / kH;                           // b*NCH + c
  const int c  = bc % NCH;
  const int b  = bc / NCH;
  const long base = ((long)b * kT + (long)c * CHUNK) * kH + d;
  float A = 0.0f, L = -INFINITY;
#pragma unroll 4
  for (int i = 0; i < CHUNK; ++i) {
    long off = base + (long)i * kH;
    A += lf[off];
    L = lse2(L, lv[off] - A);
  }
  sumA[(long)bc * kH + d] = A;
  sumL[(long)bc * kH + d] = L;
}

__global__ __launch_bounds__(256)
void scan_pass2(const float* __restrict__ sumA, const float* __restrict__ sumL,
                float* __restrict__ carA, float* __restrict__ carL)
{
  const int tid = blockIdx.x * 256 + threadIdx.x;    // B*H threads
  const int d = tid % kH;
  const int b = tid / kH;
  float A = 0.0f, L = 0.0f;                          // L0 = log h0 term (= 0)
#pragma unroll
  for (int c = 0; c < NCH; ++c) {
    long idx = ((long)b * NCH + c) * kH + d;
    carA[idx] = A;
    carL[idx] = L;
    float a = sumA[idx], l = sumL[idx];
    L = lse2(L, l - A);
    A += a;
  }
}

__global__ __launch_bounds__(256)
void scan_pass3(const float* __restrict__ lf, const float* __restrict__ lv,
                const float* __restrict__ carA, const float* __restrict__ carL,
                float* __restrict__ h)
{
  const int tid = blockIdx.x * 256 + threadIdx.x;
  const int d  = tid % kH;
  const int bc = tid / kH;
  const int c  = bc % NCH;
  const int b  = bc / NCH;
  const long base = ((long)b * kT + (long)c * CHUNK) * kH + d;
  float A = carA[(long)bc * kH + d];
  float L = carL[(long)bc * kH + d];
#pragma unroll 4
  for (int i = 0; i < CHUNK; ++i) {
    long off = base + (long)i * kH;
    A += lf[off];
    L = lse2(L, lv[off] - A);
    h[off] = __expf(A + L);
  }
}

// ---------------------------------------------------------------------------
// Head: gather last valid step, 2x (512x512) MLP + relu, 512->1 + sigmoid.
// One 512-thread block; single 32KB ping-pong LDS buffer.
// ---------------------------------------------------------------------------
__global__ __launch_bounds__(512)
void head_kernel(const float* __restrict__ h, const int* __restrict__ lengths,
                 const float* __restrict__ Wm0, const float* __restrict__ bm0,
                 const float* __restrict__ Wm1, const float* __restrict__ bm1,
                 const float* __restrict__ Wout, const float* __restrict__ bout,
                 float* __restrict__ out)
{
  __shared__ float vbuf[kB][kM];
  const int t = threadIdx.x;               // column / feature index 0..511

  // gather value[b] = h[b, max(len-1,0)], or 1.0 when len == 0
  for (int r = 0; r < kB; ++r) {
    int len = lengths[r];
    float val = 1.0f;
    if (len > 0) val = h[((long)r * kT + (len - 1)) * kH + t];
    vbuf[r][t] = val;
  }
  __syncthreads();

  float acc[kB];
  // mlp0
#pragma unroll
  for (int r = 0; r < kB; ++r) acc[r] = bm0[t];
  for (int k = 0; k < kH; ++k) {
    float w = Wm0[(long)k * kM + t];
#pragma unroll
    for (int r = 0; r < kB; ++r) acc[r] = fmaf(vbuf[r][k], w, acc[r]);
  }
  __syncthreads();
#pragma unroll
  for (int r = 0; r < kB; ++r) vbuf[r][t] = fmaxf(acc[r], 0.0f);
  __syncthreads();

  // mlp1
#pragma unroll
  for (int r = 0; r < kB; ++r) acc[r] = bm1[t];
  for (int k = 0; k < kM; ++k) {
    float w = Wm1[(long)k * kM + t];
#pragma unroll
    for (int r = 0; r < kB; ++r) acc[r] = fmaf(vbuf[r][k], w, acc[r]);
  }
  __syncthreads();
#pragma unroll
  for (int r = 0; r < kB; ++r) vbuf[r][t] = fmaxf(acc[r], 0.0f);
  __syncthreads();

  // output row per lane
  if (t < kB) {
    float s = bout[0];
    for (int k = 0; k < kM; ++k) s = fmaf(vbuf[t][k], Wout[k], s);
    out[t] = 1.0f / (1.0f + __expf(-s));
  }
}

// ---------------------------------------------------------------------------
extern "C" void kernel_launch(void* const* d_in, const int* in_sizes, int n_in,
                              void* d_out, int out_size, void* d_ws, size_t ws_size,
                              hipStream_t stream)
{
  (void)in_sizes; (void)n_in; (void)out_size; (void)ws_size;

  const int*   x    = (const int*)  d_in[0];
  const int*   len  = (const int*)  d_in[1];
  const float* emb  = (const float*)d_in[2];
  const float* Wf0  = (const float*)d_in[3];
  const float* bf0  = (const float*)d_in[4];
  const float* Wi0  = (const float*)d_in[5];
  const float* bi0  = (const float*)d_in[6];
  const float* Wh0  = (const float*)d_in[7];
  const float* bh0  = (const float*)d_in[8];
  const float* Wf1  = (const float*)d_in[9];
  const float* bf1  = (const float*)d_in[10];
  const float* Wi1  = (const float*)d_in[11];
  const float* bi1  = (const float*)d_in[12];
  const float* Wh1  = (const float*)d_in[13];
  const float* bh1  = (const float*)d_in[14];
  const float* Wm0  = (const float*)d_in[15];
  const float* bm0  = (const float*)d_in[16];
  const float* Wm1  = (const float*)d_in[17];
  const float* bm1  = (const float*)d_in[18];
  const float* Wout = (const float*)d_in[19];
  const float* bout = (const float*)d_in[20];
  float* out = (float*)d_out;

  // workspace partition
  const size_t SZ = (size_t)kN * kH;               // elements per big buffer
  char* ws = (char*)d_ws;
  float* lg_f = (float*)(ws);                      // (kN,kH)
  float* lg_v = (float*)(ws + SZ * 4);             // (kN,kH)
  float* hbuf = (float*)(ws + 2 * SZ * 4);         // (kN,kH), reused across layers
  float* sumA = (float*)(ws + 3 * SZ * 4);         // (B*NCH*H)
  float* sumL = sumA + (size_t)kB * NCH * kH;
  float* carA = sumL + (size_t)kB * NCH * kH;
  float* carL = carA + (size_t)kB * NCH * kH;

  const dim3 ggrid(kN / BM, kH / BN);              // (256, 8)
  const int p13_blocks = (kB * NCH * kH) / 256;    // 1024
  const int p2_blocks  = (kB * kH) / 256;          // 32

  // ---- layer 0 (A = emb[x[r]]) ----
  gate_gemm_kernel<<<ggrid, 256, 0, stream>>>(emb, x, Wf0, Wi0, Wh0,
                                              bf0, bi0, bh0, lg_f, lg_v);
  scan_pass1<<<p13_blocks, 256, 0, stream>>>(lg_f, lg_v, sumA, sumL);
  scan_pass2<<<p2_blocks, 256, 0, stream>>>(sumA, sumL, carA, carL);
  scan_pass3<<<p13_blocks, 256, 0, stream>>>(lg_f, lg_v, carA, carL, hbuf);

  // ---- layer 1 (A = h) ----
  gate_gemm_kernel<<<ggrid, 256, 0, stream>>>(hbuf, nullptr, Wf1, Wi1, Wh1,
                                              bf1, bi1, bh1, lg_f, lg_v);
  scan_pass1<<<p13_blocks, 256, 0, stream>>>(lg_f, lg_v, sumA, sumL);
  scan_pass2<<<p2_blocks, 256, 0, stream>>>(sumA, sumL, carA, carL);
  scan_pass3<<<p13_blocks, 256, 0, stream>>>(lg_f, lg_v, carA, carL, hbuf);

  // ---- head ----
  head_kernel<<<1, 512, 0, stream>>>(hbuf, len, Wm0, bm0, Wm1, bm1, Wout, bout, out);
}
